// EPROP3_LSTM_65326452572910
// MI455X (gfx1250) — matrix operand order repaired and checked by reading, not compiled, not run on hardware
//
#include <hip/hip_runtime.h>
#include <hip/hip_bf16.h>

#define DEV __device__ __forceinline__

typedef __attribute__((ext_vector_type(2)))  float  v2f;
typedef __attribute__((ext_vector_type(8)))  float  v8f;
typedef __attribute__((ext_vector_type(16))) __bf16 v16bf;

constexpr int BB = 32, TT = 128, II = 256, HH = 512, H3 = 1536, H4 = 2048;

DEV v8f wmma_f32(v2f a, v2f b, v8f c) {
  // V_WMMA_F32_16X16X4_F32: D = A(16x4) * B(4x16) + C
  return __builtin_amdgcn_wmma_f32_16x16x4_f32(false, a, false, b, (short)0, c, false, false);
}
DEV v8f wmma_bf16(v16bf a, v16bf b, v8f c) {
  // V_WMMA_F32_16X16X32_BF16
  return __builtin_amdgcn_wmma_f32_16x16x32_bf16(false, a, false, b, (short)0, c, false, false);
}
DEV float sigf(float x) { return 1.0f / (1.0f + __expf(-x)); }

// ---------------------------------------------------------------------------
// Pack Whh [4H,H] f32 -> bf16 B-operand tiles in exact lane layout.
// packed[(((nt*16 + kc)*32 + lane)*16 + j)] = bf16(Whh[n, K])
//   n = nt*16 + lane%16 ; K = kc*32 + (j/8)*16 + (lane/16)*8 + j%8   (B 32x16 dense layout)
// ---------------------------------------------------------------------------
__global__ void k_pack_whh(const float* __restrict__ Whh, __bf16* __restrict__ pB) {
  int gid = blockIdx.x * 256 + threadIdx.x;      // 1,048,576 entries
  int j  = gid & 15;
  int L  = (gid >> 4) & 31;
  int kc = (gid >> 9) & 15;
  int nt = gid >> 13;
  int n = nt * 16 + (L & 15);
  int K = kc * 32 + (j >> 3) * 16 + (L >> 4) * 8 + (j & 7);
  pB[gid] = (__bf16)Whh[n * HH + K];
}

// ---------------------------------------------------------------------------
// Gpre[t][b][4H] = x[b,t,:] @ Wih^T + b_lstm    (M=4096, N=2048, K=256, f32 WMMA)
// ---------------------------------------------------------------------------
__global__ void k_gemm_wih(const float* __restrict__ X, const float* __restrict__ Wih,
                           const float* __restrict__ bl, float* __restrict__ G) {
  int lane = threadIdx.x & 31, wave = threadIdx.x >> 5;
  int tile = blockIdx.x * 8 + wave;              // 32768 tiles (256 x 128)
  int mt = tile >> 7, nt = tile & 127;
  int half = lane >> 4, l16 = lane & 15;
  int m = mt * 16 + l16;                         // m = t*32 + b
  int t = m >> 5, b = m & 31;
  const float* arow = X + (b * TT + t) * II;
  int col = nt * 16 + l16;
  const float* brow = Wih + col * II;
  v8f acc = {};
  for (int k0 = 0; k0 < II; k0 += 4) {
    int kb = k0 + 2 * half;
    v2f a  = *(const v2f*)(arow + kb);
    v2f bv = *(const v2f*)(brow + kb);
    acc = wmma_f32(a, bv, acc);
  }
  float bias = bl[col];
  for (int v = 0; v < 8; ++v) {
    int mr = mt * 16 + v + 8 * half;
    int tr = mr >> 5, br = mr & 31;
    G[(tr * BB + br) * H4 + col] = acc[v] + bias;
  }
}

// ---------------------------------------------------------------------------
// Sequential LSTM recurrence. Batch rows independent -> 2 persistent WGs
// (16 rows each), no inter-WG sync. h@Whh^T in bf16 WMMA, c in registers,
// h kept in LDS as a pre-packed bf16 A-fragment buffer.
// Emits f_all [T][B][H], d_all [T][B][3H], Hs [T+1][B][H] (Hs[t] = h_{t-1}).
// ---------------------------------------------------------------------------
__global__ void __launch_bounds__(256) k_recur(const float* __restrict__ G,
                                               const __bf16* __restrict__ pB,
                                               float* __restrict__ f_all,
                                               float* __restrict__ d_all,
                                               float* __restrict__ Hs) {
  __shared__ __bf16 ldsA[16 * 32 * 16];          // 16 KB: [kc][lane][16] A-frags
  int tid = threadIdx.x, lane = tid & 31, wave = tid >> 5;
  int wg = blockIdx.x;                           // 0..1, batch rows wg*16..+15
  int half = lane >> 4, l16 = lane & 15;

  for (int e = tid; e < 8192; e += 256) ldsA[e] = (__bf16)0.0f;   // h0 = 0
  for (int e = tid; e < 16 * HH; e += 256) {                       // Hs[0] = 0
    int r = e >> 9, n = e & 511;
    Hs[(wg * 16 + r) * HH + n] = 0.0f;
  }
  float creg[4][8], hreg[4][8];
  for (int blk = 0; blk < 4; ++blk)
    for (int v = 0; v < 8; ++v) creg[blk][v] = 0.0f;
  __syncthreads();

  for (int t = 0; t < TT; ++t) {
    for (int blk = 0; blk < 4; ++blk) {
      int nb = wave * 4 + blk;                   // hidden 16-col block 0..31
      int colh = nb * 16 + l16;                  // hidden unit index 0..511
      v8f acc[4];
      for (int q = 0; q < 4; ++q)                // C := Gpre (bias + x-part)
        for (int v = 0; v < 8; ++v) {
          int bg = wg * 16 + v + 8 * half;
          acc[q][v] = G[(t * BB + bg) * H4 + q * HH + colh];
        }
      for (int kc = 0; kc < 16; ++kc) {
        v16bf a = *(const v16bf*)&ldsA[(kc * 32 + lane) * 16];
        for (int q = 0; q < 4; ++q) {
          const __bf16* bp = pB + (size_t)(((q * 32 + nb) * 16 + kc) * 32 + lane) * 16;
          v16bf bv = *(const v16bf*)bp;
          acc[q] = wmma_bf16(a, bv, acc[q]);
        }
      }
      for (int v = 0; v < 8; ++v) {
        float i = sigf(acc[0][v]);
        float f = sigf(acc[1][v]);
        float g = tanhf(acc[2][v]);
        float o = sigf(acc[3][v]);
        float cp = creg[blk][v];
        float di = g * i * (1.0f - i);
        float df = cp * f * (1.0f - f);
        float dg = i * (1.0f - g * g);
        float cn = f * cp + i * g;
        float hn = o * tanhf(cn);
        creg[blk][v] = cn;
        hreg[blk][v] = hn;
        int bg = wg * 16 + v + 8 * half;
        f_all[(t * BB + bg) * HH + colh] = f;
        float* dp = d_all + (size_t)(t * BB + bg) * H3 + colh;
        dp[0] = di; dp[HH] = df; dp[2 * HH] = dg;
        Hs[(size_t)((t + 1) * BB + bg) * HH + colh] = hn;
      }
    }
    __syncthreads();                             // all waves done reading ldsA
    // repack h_t into bf16 A-fragment layout (16-bit A 16x32 per ISA table)
    for (int blk = 0; blk < 4; ++blk) {
      int nb = wave * 4 + blk;
      int K = nb * 16 + l16;                     // K dim of next step = hidden idx
      int kc = K >> 5, ko = K & 31, q2 = ko >> 3;
      int j = (ko & 7) + 8 * (q2 >> 1);
      for (int v = 0; v < 8; ++v) {
        int row = v + 8 * half;                  // A-matrix M row (batch-local)
        int L2 = (q2 & 1) * 16 + row;
        ldsA[(kc * 32 + L2) * 16 + j] = (__bf16)hreg[blk][v];
      }
    }
    __syncthreads();
  }
}

// ---------------------------------------------------------------------------
// Backward scan: A[b][j][t] = d_t[b,j] * prod_{r>t} f_r[b, j%H];  ev_b = row sum
// ---------------------------------------------------------------------------
__global__ void k_alpha(const float* __restrict__ d_all, const float* __restrict__ f_all,
                        float* __restrict__ Atr, float* __restrict__ evb) {
  int gid = blockIdx.x * 256 + threadIdx.x;      // 49152 rows
  int b = gid / H3, j = gid % H3;
  int jh = j & 511;
  float P = 1.0f, s = 0.0f;
  float* arow = Atr + (size_t)gid * TT;
  for (int t = TT - 1; t >= 0; --t) {
    float dv = d_all[(size_t)(t * BB + b) * H3 + j];
    float al = dv * P;
    arow[t] = al;
    s += al;
    P *= f_all[(size_t)(t * BB + b) * HH + jh];
  }
  evb[gid] = s;
}

// ---------------------------------------------------------------------------
// Batched GEMMs: ev_x[b] = A[b](1536x128) @ X[b](128x256)
//                ev_h[b] = A[b]           @ H[b](128x512)   (f32 WMMA)
// ---------------------------------------------------------------------------
__global__ void k_ev(const float* __restrict__ Atr, const float* __restrict__ X,
                     const float* __restrict__ Hs, float* __restrict__ evx,
                     float* __restrict__ evh) {
  int lane = threadIdx.x & 31, wave = threadIdx.x >> 5;
  int tile = blockIdx.x * 8 + wave;              // 49152 + 98304 tiles
  int half = lane >> 4, l16 = lane & 15;
  int b, mt, nt, N, bstride;
  const float* Bbase;
  float* out;
  if (tile < 49152) {
    b = tile / 1536; int r = tile % 1536; mt = r >> 4; nt = r & 15;
    N = II; Bbase = X + (size_t)b * TT * II; bstride = II;
    out = evx + (size_t)b * H3 * II;
  } else {
    int t2 = tile - 49152;
    b = t2 / 3072; int r = t2 % 3072; mt = r >> 5; nt = r & 31;
    N = HH; Bbase = Hs + (size_t)b * HH; bstride = BB * HH;
    out = evh + (size_t)b * H3 * HH;
  }
  int row = mt * 16 + l16, col = nt * 16 + l16;
  const float* arow = Atr + ((size_t)b * H3 + row) * TT;
  v8f acc = {};
  for (int k0 = 0; k0 < TT; k0 += 4) {
    int kb = k0 + 2 * half;
    v2f a = *(const v2f*)(arow + kb);
    v2f bv;
    bv.x = Bbase[(size_t)kb * bstride + col];
    bv.y = Bbase[(size_t)(kb + 1) * bstride + col];
    acc = wmma_f32(a, bv, acc);
  }
  for (int v = 0; v < 8; ++v)
    out[(size_t)(mt * 16 + v + 8 * half) * N + col] = acc[v];
}

// ---------------------------------------------------------------------------
// Generic dense: out[M,N] = act( A[M,K] @ W[N,K]^T + bias ), f32 WMMA, N-guarded
// ---------------------------------------------------------------------------
__global__ void k_dense(const float* __restrict__ A, const float* __restrict__ W,
                        const float* __restrict__ bias, float* __restrict__ out,
                        int M, int N, int K, int relu) {
  int lane = threadIdx.x & 31, wave = threadIdx.x >> 5;
  int tiles_n = (N + 15) >> 4;
  int tiles = ((M + 15) >> 4) * tiles_n;
  int tile = blockIdx.x * 8 + wave;
  if (tile >= tiles) return;                     // wave-uniform
  int mt = tile / tiles_n, nt = tile % tiles_n;
  int half = lane >> 4, l16 = lane & 15;
  int row = mt * 16 + l16; if (row >= M) row = M - 1;
  int col = nt * 16 + l16;
  bool cok = col < N;
  const float* arow = A + (size_t)row * K;
  const float* brow = W + (size_t)(cok ? col : 0) * K;
  v8f acc = {};
  for (int k0 = 0; k0 < K; k0 += 4) {
    int kb = k0 + 2 * half;
    v2f a = *(const v2f*)(arow + kb);
    v2f bv;
    if (cok) bv = *(const v2f*)(brow + kb);
    else { bv.x = 0.0f; bv.y = 0.0f; }
    acc = wmma_f32(a, bv, acc);
  }
  float bi = cok ? bias[col] : 0.0f;
  for (int v = 0; v < 8; ++v) {
    int r = mt * 16 + v + 8 * half;
    if (r < M && cok) {
      float val = acc[v] + bi;
      if (relu) val = fmaxf(val, 0.0f);
      out[(size_t)r * N + col] = val;
    }
  }
}

extern "C" void kernel_launch(void* const* d_in, const int* in_sizes, int n_in,
                              void* d_out, int out_size, void* d_ws, size_t ws_size,
                              hipStream_t stream) {
  const float* x   = (const float*)d_in[0];
  const float* Wih = (const float*)d_in[1];
  const float* Whh = (const float*)d_in[2];
  const float* bl  = (const float*)d_in[3];
  const float* Wd  = (const float*)d_in[4];
  const float* bd  = (const float*)d_in[5];
  const float* Ws1 = (const float*)d_in[6];
  const float* bs1 = (const float*)d_in[7];
  const float* Ws2 = (const float*)d_in[8];
  const float* bs2 = (const float*)d_in[9];
  const float* Ws3 = (const float*)d_in[10];
  const float* bs3 = (const float*)d_in[11];
  const float* Ws4 = (const float*)d_in[12];
  const float* bs4 = (const float*)d_in[13];

  char* ws = (char*)d_ws;
  float*  G     = (float*)(ws + 0);              // 33.55 MB  [T][B][4H]
  __bf16* pB    = (__bf16*)(ws + 33554432);      //  2.10 MB  packed Whh^T bf16
  float*  f_all = (float*)(ws + 35651584);       //  8.39 MB  [T][B][H]
  float*  d_all = (float*)(ws + 44040192);       // 25.17 MB  [T][B][3H]
  float*  Hs    = (float*)(ws + 69206016);       //  8.45 MB  [T+1][B][H]
  float*  Atr   = (float*)(ws + 77660160);       // 25.17 MB  [B][3H][T]
  float*  z1    = (float*)(ws + 102825984);      // 64 KB
  float*  z2    = (float*)(ws + 102891520);      // 32 KB
  float*  z3    = (float*)(ws + 102924288);      // 16 KB

  float* o    = (float*)d_out;
  float* pred = o;                               // [32,10]
  float* sg   = o + 320;                         // [32,512]
  float* evx  = o + 16704;                       // [32,1536,256]
  float* evh  = o + 12599616;                    // [32,1536,512]
  float* evb  = o + 37765440;                    // [32,1536,1]
  const float* last = Hs + (size_t)TT * BB * HH; // h_T

  hipLaunchKernelGGL(k_pack_whh, dim3(4096), dim3(256), 0, stream, Whh, pB);
  hipLaunchKernelGGL(k_gemm_wih, dim3(4096), dim3(256), 0, stream, x, Wih, bl, G);
  hipLaunchKernelGGL(k_recur, dim3(2), dim3(256), 0, stream, G, pB, f_all, d_all, Hs);
  hipLaunchKernelGGL(k_alpha, dim3(192), dim3(256), 0, stream, d_all, f_all, Atr, evb);
  hipLaunchKernelGGL(k_ev, dim3(18432), dim3(256), 0, stream, Atr, x, Hs, evx, evh);
  hipLaunchKernelGGL(k_dense, dim3(8), dim3(256), 0, stream, last, Ws1, bs1, z1, 32, 512, 512, 1);
  hipLaunchKernelGGL(k_dense, dim3(4), dim3(256), 0, stream, z1, Ws2, bs2, z2, 32, 256, 512, 1);
  hipLaunchKernelGGL(k_dense, dim3(2), dim3(256), 0, stream, z2, Ws3, bs3, z3, 32, 128, 256, 1);
  hipLaunchKernelGGL(k_dense, dim3(8), dim3(256), 0, stream, z3, Ws4, bs4, sg, 32, 512, 128, 0);
  hipLaunchKernelGGL(k_dense, dim3(1), dim3(256), 0, stream, last, Wd, bd, pred, 32, 10, 512, 0);
}
